// R3MHFAdapterLM_19662360281121
// MI455X (gfx1250) — matrix operand order repaired and checked by reading, not compiled, not run on hardware
//
#include <hip/hip_runtime.h>
#include <hip/hip_bf16.h>

#define BB 8
#define TT 4096
#define DD 896
#define NSLOTS 64
#define NTHREADS 896          // 28 waves, one thread per feature dim
#define NWAVES 28
#define GATE_MAX 0.2f
#define MEM_SCALE 0.5f
#define DELTA_CLIP 2.0f
#define LN_EPS 1e-5f

// LDS row stride (in dwords) of the TDM-padded W stripe: 448 data dwords per
// row + 1 pad dword per 64 data dwords = 455.  455 % 64 = 7 (coprime with the
// 64 LDS banks) -> near-conflict-free ds_load_b128 fragment reads.
#define BROW_DW 455

typedef __attribute__((ext_vector_type(16))) __bf16 v16bf;
typedef __attribute__((ext_vector_type(8)))  float  v8f;
typedef __attribute__((ext_vector_type(4)))  unsigned int u32x4;
typedef __attribute__((ext_vector_type(4)))  int i32x4;
typedef __attribute__((ext_vector_type(8)))  int i32x8;

__device__ __forceinline__ float wave_sum(float v) {
#pragma unroll
  for (int off = 16; off > 0; off >>= 1) v += __shfl_xor(v, off, 32);
  return v;
}

__device__ __forceinline__ unsigned short f2bf(float x) {
  unsigned int u = __float_as_uint(x);
  u += 0x7FFFu + ((u >> 16) & 1u);          // round-to-nearest-even
  return (unsigned short)(u >> 16);
}

// ---------------------------------------------------------------------------
// Phase 1: sequential memory scan. One workgroup per batch; the full slot
// memory (64x896 f32 = 229KB) lives in LDS for all 4096 steps.  h_t and the
// attention mask are software-pipelined one step ahead so no load latency
// sits on the per-step critical path.
// ---------------------------------------------------------------------------
__global__ void __launch_bounds__(NTHREADS)
scan_kernel(const float* __restrict__ h, const int* __restrict__ mask,
            const float* __restrict__ ln_g, const float* __restrict__ ln_b,
            const float* __restrict__ W_nm, const float* __restrict__ b_nm,
            const float* __restrict__ W_wg, const float* __restrict__ b_wg,
            unsigned short* __restrict__ MN)
{
  extern __shared__ float smem[];
  float* mem    = smem;                     // 64*896
  float* hbuf   = mem + NSLOTS * DD;        // 896
  float* scores = hbuf + DD;                // 64
  float* attn   = scores + NSLOTS;          // 64
  float* wred   = attn + NSLOTS;            // NWAVES*4
  float* scal   = wred + NWAVES * 4;        // 8 scalars: wg, mu, rstd, idx-bits
  float* wnm_s  = scal + 8;                 // 2*896
  float* wwg_s  = wnm_s + 2 * DD;           // 897
  float* lng_s  = wwg_s + (DD + 1);         // 896
  float* lnb_s  = lng_s + DD;               // 896

  const int b    = blockIdx.x;
  const int tid  = threadIdx.x;
  const int lane = tid & 31;
  const int wave = tid >> 5;

  for (int i = tid; i < NSLOTS * DD; i += NTHREADS) mem[i] = 0.f;
  for (int i = tid; i < 2 * DD;      i += NTHREADS) wnm_s[i] = W_nm[i];
  for (int i = tid; i < DD + 1;      i += NTHREADS) wwg_s[i] = W_wg[i];
  lng_s[tid] = ln_g[tid];
  lnb_s[tid] = ln_b[tid];
  __syncthreads();

  const float scale = rsqrtf((float)DD);
  const float bnm = b_nm[0], bwg = b_wg[0];
  const float* hb = h + (size_t)b * TT * DD;
  unsigned short* mnb = MN + (size_t)b * TT * DD;
  const int* maskb = mask + b * TT;

  // software pipeline: h_{t+1} / keep_{t+1} loads issued a full step early
  float hd_next = hb[tid];
  float kp_next = (float)maskb[0];

  for (int t = 0; t < TT; ++t) {
    const float hd = hd_next;
    const float keep = kp_next;
    if (t + 1 < TT) {
      hd_next = hb[(size_t)(t + 1) * DD + tid];
      kp_next = (float)maskb[t + 1];
      __builtin_prefetch(&hb[(size_t)(t + 2) * DD + tid], 0, 3);
    }
    hbuf[tid] = hd;
    __syncthreads();

    // scores[s] = scale * dot(h_t, mem[s]) ; waves stride over the 64 slots
    for (int s = wave; s < NSLOTS; s += NWAVES) {
      const float4* mrow = (const float4*)(mem + s * DD);
      const float4* hrow = (const float4*)hbuf;
      float acc = 0.f;
#pragma unroll
      for (int it = 0; it < DD / 128; ++it) {
        float4 mv = mrow[it * 32 + lane];
        float4 hv = hrow[it * 32 + lane];
        acc += mv.x * hv.x + mv.y * hv.y + mv.z * hv.z + mv.w * hv.w;
      }
      acc = wave_sum(acc);
      if (lane == 0) scores[s] = acc * scale;
    }
    __syncthreads();

    // softmax over 64 slots + argmax (first-max tie-break), by wave 0
    if (wave == 0) {
      float s0 = scores[lane], s1 = scores[lane + 32];
      float mval; int midx;
      if (s1 > s0) { mval = s1; midx = lane + 32; } else { mval = s0; midx = lane; }
      float gmax = mval;
#pragma unroll
      for (int off = 16; off > 0; off >>= 1)
        gmax = fmaxf(gmax, __shfl_xor(gmax, off, 32));
#pragma unroll
      for (int off = 16; off > 0; off >>= 1) {
        float ov = __shfl_xor(mval, off, 32);
        int   oi = __shfl_xor(midx, off, 32);
        if (ov > mval || (ov == mval && oi < midx)) { mval = ov; midx = oi; }
      }
      float e0 = __expf(s0 - gmax), e1 = __expf(s1 - gmax);
      float inv = 1.f / wave_sum(e0 + e1);
      attn[lane]      = e0 * inv;
      attn[lane + 32] = e1 * inv;
      if (lane == 0) ((int*)scal)[3] = midx;
    }
    __syncthreads();

    // m[d] = sum_s attn[s]*mem[s][d]  (one d per thread) + reduction partials
    float md = 0.f;
#pragma unroll 8
    for (int s = 0; s < NSLOTS; ++s) md += attn[s] * mem[s * DD + tid];
    float pg  = hd * wnm_s[tid] + md * wnm_s[DD + tid];
    float pw  = hd * wwg_s[tid];
    float pm  = md;
    float pm2 = md * md;
    pg = wave_sum(pg); pw = wave_sum(pw); pm = wave_sum(pm); pm2 = wave_sum(pm2);
    if (lane == 0) {
      wred[wave * 4 + 0] = pg; wred[wave * 4 + 1] = pw;
      wred[wave * 4 + 2] = pm; wred[wave * 4 + 3] = pm2;
    }
    __syncthreads();

    // final reduction + gate scalars: parallel in wave 0 (off critical path)
    if (wave == 0) {
      float rg  = (lane < NWAVES) ? wred[lane * 4 + 0] : 0.f;
      float rw  = (lane < NWAVES) ? wred[lane * 4 + 1] : 0.f;
      float rm  = (lane < NWAVES) ? wred[lane * 4 + 2] : 0.f;
      float rm2 = (lane < NWAVES) ? wred[lane * 4 + 3] : 0.f;
      float Pg = wave_sum(rg), Pw = wave_sum(rw);
      float Pm = wave_sum(rm), Pm2 = wave_sum(rm2);
      if (lane == 0) {
        float g  = 1.f / (1.f + __expf(-(Pg + bnm)));
        float wg = 1.f / (1.f + __expf(-(Pw + wwg_s[DD] * g + bwg)));
        wg = fminf(fmaxf(wg, 0.f), GATE_MAX) * keep;
        float mu  = Pm * (1.f / DD);
        float var = Pm2 * (1.f / DD) - mu * mu;
        scal[0] = wg; scal[1] = mu; scal[2] = rsqrtf(var + LN_EPS);
      }
    }
    __syncthreads();

    const float wg = scal[0], mu = scal[1], rstd = scal[2];
    const int idx = ((int*)scal)[3];
    // top-1 hard write (forward value of straight-through selection)
    float* mrow = mem + idx * DD;
    mrow[tid] = mrow[tid] * (1.f - wg) + wg * hd;
    // LayerNorm(m) -> bf16 for the phase-2 WMMA GEMM
    float mn = (md - mu) * rstd * lng_s[tid] + lnb_s[tid];
    mnb[(size_t)t * DD + tid] = f2bf(mn);
    __syncthreads();
  }
}

// ---------------------------------------------------------------------------
// W_out f32 -> bf16 (L2-resident, 1.6MB)
// ---------------------------------------------------------------------------
__global__ void convert_w(const float* __restrict__ W, unsigned short* __restrict__ Wb, int n) {
  int i = blockIdx.x * blockDim.x + threadIdx.x;
  if (i < n) Wb[i] = f2bf(W[i]);
}

// ---------------------------------------------------------------------------
// Phase 2: out = h + clip(0.5 * (mn @ W_out^T), +-2).
// Block = 8 waves (4 row x 2 col), each wave computes a 32x32 macro-tile with
// v_wmma_f32_16x16x32_bf16.  The block's 64-column W stripe (64x896 bf16) is
// DMA'd into LDS once by the Tensor Data Mover with 1-dword-per-64 padding,
// then all fragment B reads come from LDS.
// ---------------------------------------------------------------------------
__device__ __forceinline__ v8f wmma_bf16(const v16bf& a, const v16bf& b, const v8f& c) {
  return __builtin_amdgcn_wmma_f32_16x16x32_bf16(false, a, false, b, (short)0, c, false, false);
}

__global__ void __launch_bounds__(256)
gemm_kernel(const unsigned short* __restrict__ MN, const unsigned short* __restrict__ Wb,
            const float* __restrict__ h, float* __restrict__ out)
{
  extern __shared__ unsigned short bsh[];   // padded W stripe: 64 rows x 455 dwords

  const int lane = threadIdx.x & 31;
  const int wave = threadIdx.x >> 5;
  const int r    = lane & 15;
  const int half = lane >> 4;

  // --- TDM: load W stripe [64 x 896] bf16 -> LDS (wave 0 issues, all wait) ---
  if (threadIdx.x < 32) {
    unsigned long long gaddr =
        (unsigned long long)(uintptr_t)(Wb + (size_t)blockIdx.x * 64 * DD);
    unsigned int lds_lo = (unsigned int)(uintptr_t)(&bsh[0]);  // low 32b = LDS offset

    u32x4 g0;
    g0[0] = 1u;                                                   // count=1
    g0[1] = (unsigned int)__builtin_amdgcn_readfirstlane((int)lds_lo);
    g0[2] = (unsigned int)__builtin_amdgcn_readfirstlane((int)(unsigned int)gaddr);
    g0[3] = ((unsigned int)__builtin_amdgcn_readfirstlane(
                 (int)(unsigned int)((gaddr >> 32) & 0x01FFFFFFull)))
            | 0x80000000u;                                        // type=2 (image)

    i32x8 g1;
    g1[0] = (1 << 16) | (1 << 20) | (5 << 22);  // data_size=2B, pad_en, intvl=64dw, amt=1dw
    g1[1] = (DD & 0xFFFF) << 16;                // tensor_dim0[15:0]
    g1[2] = (64 << 16);                         // tensor_dim0 hi=0 | tensor_dim1=64
    g1[3] = (DD << 16);                         // tensor_dim1 hi=0 | tile_dim0=896
    g1[4] = 64;                                 // tile_dim1=64, tile_dim2=0
    g1[5] = DD;                                 // tensor_dim0_stride
    g1[6] = 0; g1[7] = 0;

    i32x4 gz = {0, 0, 0, 0};
#if __clang_major__ >= 23
    i32x8 gz8 = {0, 0, 0, 0, 0, 0, 0, 0};
    __builtin_amdgcn_tensor_load_to_lds(g0, g1, gz, gz, gz8, 0);
#else
    __builtin_amdgcn_tensor_load_to_lds(g0, g1, gz, gz, 0);
#endif
    __builtin_amdgcn_s_wait_tensorcnt(0);
  }
  __syncthreads();

  const int wr = wave >> 1;                          // 0..3
  const int wc = wave & 1;                           // 0..1
  const int row0  = blockIdx.y * 128 + wr * 32;      // 2 x 16-row tiles
  const int col0  = blockIdx.x * 64  + wc * 32;      // 2 x 16-col tiles (global)
  const int nloc0 = wc * 32;                         // local col base in LDS stripe

  const unsigned short* a0row = MN + (size_t)(row0 + r) * DD;
  const unsigned short* a1row = MN + (size_t)(row0 + 16 + r) * DD;
  const unsigned int nl0 = (unsigned int)(nloc0 + r);
  const unsigned int nl1 = (unsigned int)(nloc0 + 16 + r);

  v8f c00 = {}, c01 = {}, c10 = {}, c11 = {};
  for (int kk = 0; kk < DD; kk += 32) {
    union { unsigned int u[8]; v16bf v; } a0, a1, b0, b1;
    const int j = (kk >> 1) + half * 8;              // dword offset in unpadded row
    const unsigned int ldsdw = (unsigned int)(j + (j >> 6));  // +1 pad dword per 64
#pragma unroll
    for (int v = 0; v < 8; ++v) {
      // A 16x32 bf16 per-lane layout: K = half*16 + v*2, pairs packed
      const int ka = kk + half * 16 + v * 2;
      a0.u[v] = *(const unsigned int*)(a0row + ka);
      a1.u[v] = *(const unsigned int*)(a1row + ka);
      // B 32x16 bf16: lanes 0-15 hold K=0..15, lanes 16-31 hold K=16..31 (from LDS)
      b0.u[v] = *(const unsigned int*)((const char*)bsh + (size_t)(nl0 * BROW_DW + ldsdw + v) * 4u);
      b1.u[v] = *(const unsigned int*)((const char*)bsh + (size_t)(nl1 * BROW_DW + ldsdw + v) * 4u);
    }
    c00 = wmma_bf16(a0.v, b0.v, c00);
    c01 = wmma_bf16(a0.v, b1.v, c01);
    c10 = wmma_bf16(a1.v, b0.v, c10);
    c11 = wmma_bf16(a1.v, b1.v, c11);
  }

  auto emit = [&](const v8f& c, int rbase, int cbase) {
#pragma unroll
    for (int v = 0; v < 8; ++v) {
      size_t o = (size_t)(rbase + v + 8 * half) * DD + (size_t)(cbase + r);
      float d = c[v] * MEM_SCALE;
      d = fminf(fmaxf(d, -DELTA_CLIP), DELTA_CLIP);
      out[o] = h[o] + d;
    }
  };
  emit(c00, row0,      col0);
  emit(c01, row0,      col0 + 16);
  emit(c10, row0 + 16, col0);
  emit(c11, row0 + 16, col0 + 16);
}

extern "C" void kernel_launch(void* const* d_in, const int* in_sizes, int n_in,
                              void* d_out, int out_size, void* d_ws, size_t ws_size,
                              hipStream_t stream) {
  (void)in_sizes; (void)n_in; (void)out_size; (void)ws_size;
  const float* h     = (const float*)d_in[0];
  const int*   mask  = (const int*)d_in[1];
  const float* ln_g  = (const float*)d_in[2];
  const float* ln_b  = (const float*)d_in[3];
  const float* W_out = (const float*)d_in[4];
  const float* W_nm  = (const float*)d_in[5];
  const float* b_nm  = (const float*)d_in[6];
  const float* W_wg  = (const float*)d_in[7];
  const float* b_wg  = (const float*)d_in[8];
  float* out = (float*)d_out;

  unsigned short* MN = (unsigned short*)d_ws;                 // [B*T, D] bf16
  unsigned short* Wb = MN + (size_t)BB * TT * DD;             // [D, D]   bf16

  const size_t scan_lds =
      (size_t)(NSLOTS * DD + DD + NSLOTS + NSLOTS + NWAVES * 4 + 8 +
               2 * DD + (DD + 1) + DD + DD) * sizeof(float);  // ~252 KB (of 320 KB/WGP)
  const size_t gemm_lds = (size_t)64 * BROW_DW * 4;           // padded W stripe, ~114 KB

  convert_w<<<(DD * DD + 255) / 256, 256, 0, stream>>>(W_out, Wb, DD * DD);
  scan_kernel<<<BB, NTHREADS, scan_lds, stream>>>(h, mask, ln_g, ln_b,
                                                  W_nm, b_nm, W_wg, b_wg, MN);
  gemm_kernel<<<dim3(DD / 64, (BB * TT) / 128), 256, 0, stream>>>(MN, Wb, h, out);
}
